// AgentAwareAttentionV2_10771777978766
// MI455X (gfx1250) — matrix operand order repaired and checked by reading, not compiled, not run on hardware
//
#include <hip/hip_runtime.h>
#include <hip/hip_bf16.h>
#include <cstdint>

// ---------------------------------------------------------------------------
// Problem constants (fixed by the reference)
// ---------------------------------------------------------------------------
static constexpr int Bc = 8, Lc = 1024, Sc = 1024, Dc = 256, Vc = 256, Hc = 8;
static constexpr int HDc = Dc / Hc;                       // 32
static constexpr float SCALEc = 0.17677669529663687f;     // 32^-0.5
static constexpr int ROWP  = Sc + 4;                      // f32 score row pad (bank-skewed)
static constexpr int ROWPB = Sc + 16;                     // bf16 prob row pad (16B-aligned rows)

typedef __bf16 v16bf __attribute__((ext_vector_type(16)));
typedef float  v8f   __attribute__((ext_vector_type(8)));

union Frag16 {
    v16bf v;
    unsigned short u[16];
    uint4 q[2];
};

__device__ __forceinline__ unsigned short f2bf(float f) {
    unsigned int x = __float_as_uint(f);
    return (unsigned short)((x + 0x7FFFu + ((x >> 16) & 1u)) >> 16);  // RNE
}

__device__ __forceinline__ v8f wmma_bf16(v16bf a, v16bf b, v8f c) {
    // D = A(16x32 bf16) * B(32x16 bf16) + C(16x16 f32)
    return __builtin_amdgcn_wmma_f32_16x16x32_bf16(
        /*neg_a=*/false, a, /*neg_b=*/false, b,
        /*c_mod=*/(short)0, c, /*reuse_a=*/false, /*reuse_b=*/false);
}

// ---------------------------------------------------------------------------
// Zero-fill (attention-sum accumulator must start at 0 each call)
// ---------------------------------------------------------------------------
__global__ void zero_f32(float* __restrict__ p, int n) {
    int i = blockIdx.x * blockDim.x + threadIdx.x;
    if (i < n) p[i] = 0.0f;
}

// ---------------------------------------------------------------------------
// One-pass fp32 -> bf16 conversion (4 elements/thread, b128 in / b64 out)
// ---------------------------------------------------------------------------
__global__ void cvt_bf16(const float* __restrict__ src, unsigned short* __restrict__ dst,
                         int n4) {
    int i = blockIdx.x * blockDim.x + threadIdx.x;
    if (i >= n4) return;
    float4 f = ((const float4*)src)[i];
    uint2 o;
    o.x = (unsigned)f2bf(f.x) | ((unsigned)f2bf(f.y) << 16);
    o.y = (unsigned)f2bf(f.z) | ((unsigned)f2bf(f.w) << 16);
    ((uint2*)dst)[i] = o;
}

// ---------------------------------------------------------------------------
// bf16 GEMM: Y[t,o] = sum_d X[t,d] * W[o,d]    (torch x @ W.T)
// One wave computes a 16(M) x 64(N) strip: A fragments loaded once (8 frags),
// B streamed; 32 x v_wmma_f32_16x16x32_bf16 per wave, zero VALU conversion
// in the K loop (inputs pre-converted to bf16).
// MODE 0: bf16 out [T,256], scaled          (q/k projections)
// MODE 1: bf16 out transposed [B,H,32,S]    (v projection)
// MODE 2: f32 out [T,256] + bias            (final fc)
// ---------------------------------------------------------------------------
template <int MODE>
__global__ void gemm_bf(const unsigned short* __restrict__ X,
                        const unsigned short* __restrict__ W,
                        float scale, const float* __restrict__ bias,
                        void* __restrict__ Y) {
    const int lane = threadIdx.x;
    const int hi = lane >> 4, ln = lane & 15;
    const int t0 = blockIdx.x * 16, ob = blockIdx.y * 64;

    // A fragments for the whole K=256 sweep (layout: k = e + 8*hi, +16 chunk)
    Frag16 afr[8];
    {
        const unsigned short* xr = X + (size_t)(t0 + ln) * Dc;
#pragma unroll
        for (int ks = 0; ks < 8; ++ks) {
            afr[ks].q[0] = *(const uint4*)(xr + ks * 32 + 8 * hi);
            afr[ks].q[1] = *(const uint4*)(xr + ks * 32 + 16 + 8 * hi);
        }
    }

#pragma unroll
    for (int ot = 0; ot < 4; ++ot) {
        const int o0 = ob + ot * 16;
        const unsigned short* wr = W + (size_t)(o0 + ln) * Dc + 8 * hi;   // B: N=ln, k=e+8*hi, +16 chunk
        v8f acc = {};
#pragma unroll
        for (int ks = 0; ks < 8; ++ks) {
            Frag16 b;
            b.q[0] = ((const uint4*)(wr + ks * 32))[0];
            b.q[1] = ((const uint4*)(wr + ks * 32 + 16))[0];
            acc = wmma_bf16(afr[ks].v, b.v, acc);
        }
#pragma unroll
        for (int r = 0; r < 8; ++r) {
            const int m = r + 8 * hi;                   // C row
            const int t = t0 + m, o = o0 + ln;
            if (MODE == 0) {
                ((unsigned short*)Y)[(size_t)t * Dc + o] = f2bf(acc[r] * scale);
            } else if (MODE == 1) {
                const int b_ = t >> 10, s = t & (Sc - 1);   // T = B*S, S=1024
                const int h = o >> 5, ol = o & 31;
                ((unsigned short*)Y)[(((size_t)b_ * Hc + h) * HDc + ol) * Sc + s] =
                    f2bf(acc[r]);
            } else {
                ((float*)Y)[(size_t)t * Vc + o] = acc[r] + bias[o];
            }
        }
    }
}

// ---------------------------------------------------------------------------
// Fused agent-aware attention for one (b, h, 16-query-row tile).
// 64 s-tiles x 2 WMMA (self/other scores) -> identity select + mask -> LDS
// -> exact softmax (wave32 shfl reductions), probabilities stored bf16 in LDS
// -> atomic head-sum output -> PV: 64 WMMA against transposed bf16 V.
// ---------------------------------------------------------------------------
__global__ void attn_kernel(const unsigned short* __restrict__ qs,
                            const unsigned short* __restrict__ qo,
                            const unsigned short* __restrict__ ksf,
                            const unsigned short* __restrict__ kof,
                            const unsigned short* __restrict__ vt,
                            const int* __restrict__ qid,
                            const int* __restrict__ kid,
                            const float* __restrict__ mask,
                            unsigned short* __restrict__ hidden,
                            float* __restrict__ attn_sum) {
    extern __shared__ char smem_raw[];
    float* sc = (float*)smem_raw;                                     // [16][ROWP] f32
    unsigned short* pb =
        (unsigned short*)(smem_raw + (size_t)16 * ROWP * sizeof(float)); // [16][ROWPB] bf16
    int* qid_sh = (int*)(smem_raw + (size_t)16 * ROWP * sizeof(float) +
                         (size_t)16 * ROWPB * sizeof(unsigned short));

    const int lane = threadIdx.x;
    const int hi = lane >> 4, ln = lane & 15;
    const int l0 = blockIdx.x * 16, h = blockIdx.y, b = blockIdx.z;

    if (lane < 16) qid_sh[lane] = qid[b * Lc + l0 + lane];
    __syncthreads();

    // Q fragments: A-matrix 16x32 bf16, head dim == K == 32 exactly.
    Frag16 aqs, aqo;
    {
        const size_t qoff = ((size_t)(b * Lc + l0 + ln)) * Dc + h * HDc;
        const unsigned short* pqs = qs + qoff;
        const unsigned short* pqo = qo + qoff;
        aqs.q[0] = *(const uint4*)(pqs + 8 * hi);
        aqs.q[1] = *(const uint4*)(pqs + 16 + 8 * hi);
        aqo.q[0] = *(const uint4*)(pqo + 8 * hi);
        aqo.q[1] = *(const uint4*)(pqo + 16 + 8 * hi);
    }

    // ---- scores: S(l,s) = select(id match, q_self.k_self, q_other.k_other) + mask
    for (int st = 0; st < Sc / 16; ++st) {
        const int s = st * 16 + ln;                 // B-matrix N = ln
        const size_t koff = ((size_t)(b * Sc + s)) * Dc + h * HDc + 8 * hi;
        Frag16 bks, bko;
        bks.q[0] = ((const uint4*)(ksf + koff))[0];
        bks.q[1] = ((const uint4*)(ksf + koff + 16))[0];
        bko.q[0] = ((const uint4*)(kof + koff))[0];
        bko.q[1] = ((const uint4*)(kof + koff + 16))[0];

        v8f cs = {}, co = {};
        cs = wmma_bf16(aqs.v, bks.v, cs);
        co = wmma_bf16(aqo.v, bko.v, co);

        const int kv = kid[b * Sc + s];
        if (st + 1 < Sc / 16)
            __builtin_prefetch(mask + ((size_t)b * Lc + l0) * Sc + s + 16, 0, 3);
#pragma unroll
        for (int r = 0; r < 8; ++r) {
            const int m = r + 8 * hi;
            const float mv = mask[((size_t)b * Lc + l0 + m) * Sc + s];
            const float val = (qid_sh[m] == kv ? cs[r] : co[r]) + mv;
            sc[m * ROWP + s] = val;
        }
    }
    __syncthreads();

    // ---- exact softmax per row; bf16 probabilities into LDS; atomic head-sum
    const float invH = 1.0f / Hc;
    for (int m = 0; m < 16; ++m) {
        float* row = sc + m * ROWP;
        unsigned short* prow = pb + m * ROWPB;
        float mx = -3.4e38f;
        for (int c = lane; c < Sc; c += 32) mx = fmaxf(mx, row[c]);
#pragma unroll
        for (int off = 16; off; off >>= 1) mx = fmaxf(mx, __shfl_xor(mx, off, 32));
        float sum = 0.0f;
        for (int c = lane; c < Sc; c += 32) {
            float e = __expf(row[c] - mx);
            row[c] = e;
            sum += e;
        }
#pragma unroll
        for (int off = 16; off; off >>= 1) sum += __shfl_xor(sum, off, 32);
        const float inv = 1.0f / sum;
        float* arow = attn_sum + ((size_t)b * Lc + l0 + m) * Sc;
        for (int c = lane; c < Sc; c += 32) {
            const float p = row[c] * inv;
            prow[c] = f2bf(p);                        // convert once, reuse twice
            unsafeAtomicAdd(&arow[c], p * invH);      // global_atomic_add_f32
        }
    }
    __syncthreads();

    // ---- PV: out[l, vd] = sum_s P[l,s] * Vh[s, vd], K swept in 32-chunks
#pragma unroll
    for (int nt = 0; nt < 2; ++nt) {                  // vd 0..15, 16..31
        v8f acc = {};
        const unsigned short* vrow =
            vt + (((size_t)(b * Hc + h)) * HDc + nt * 16 + ln) * Sc;  // N = ln
        const unsigned short* prow = pb + ln * ROWPB; // A row M = ln (bf16 in LDS)
        for (int kk = 0; kk < Sc / 32; ++kk) {
            const int sb = kk * 32;
            Frag16 pa, vb;
            pa.q[0] = *(const uint4*)(prow + sb + 8 * hi);        // ds_load_b128
            pa.q[1] = *(const uint4*)(prow + sb + 16 + 8 * hi);
            vb.q[0] = ((const uint4*)(vrow + sb + 8 * hi))[0];
            vb.q[1] = ((const uint4*)(vrow + sb + 16 + 8 * hi))[0];
            acc = wmma_bf16(pa.v, vb.v, acc);
        }
#pragma unroll
        for (int r = 0; r < 8; ++r) {
            const int m = r + 8 * hi;
            hidden[((size_t)b * Lc + l0 + m) * Vc + h * HDc + nt * 16 + ln] =
                f2bf(acc[r]);
        }
    }
}

// ---------------------------------------------------------------------------
// Host launcher
// ---------------------------------------------------------------------------
extern "C" void kernel_launch(void* const* d_in, const int* in_sizes, int n_in,
                              void* d_out, int out_size, void* d_ws, size_t ws_size,
                              hipStream_t stream) {
    (void)in_sizes; (void)n_in; (void)out_size; (void)ws_size;

    const float* q    = (const float*)d_in[0];
    const float* k    = (const float*)d_in[1];
    const float* v    = (const float*)d_in[2];
    const int*   qid  = (const int*)d_in[3];
    const int*   kid  = (const int*)d_in[4];
    const float* mask = (const float*)d_in[5];
    const float* wqs  = (const float*)d_in[6];
    const float* wqo  = (const float*)d_in[7];
    const float* wks  = (const float*)d_in[8];
    const float* wko  = (const float*)d_in[9];
    const float* wv   = (const float*)d_in[10];
    const float* fcw  = (const float*)d_in[11];
    const float* fcb  = (const float*)d_in[12];

    // ---- workspace layout (~37 MiB total), all bf16 unless noted
    const size_t nTok = (size_t)Bc * Lc;                 // 8192
    const size_t actN = nTok * Dc;                       // 2,097,152 elements
    const size_t wN   = (size_t)Dc * Dc;                 // 65,536 elements
    unsigned short* p = (unsigned short*)d_ws;
    unsigned short* q_bf   = p; p += actN;               // inputs
    unsigned short* k_bf   = p; p += actN;
    unsigned short* v_bf   = p; p += actN;
    unsigned short* wqs_bf = p; p += wN;                 // weights
    unsigned short* wqo_bf = p; p += wN;
    unsigned short* wks_bf = p; p += wN;
    unsigned short* wko_bf = p; p += wN;
    unsigned short* wv_bf  = p; p += wN;
    unsigned short* fcw_bf = p; p += wN;
    unsigned short* qs_bf  = p; p += actN;               // projections
    unsigned short* qo_bf  = p; p += actN;
    unsigned short* ks_bf  = p; p += actN;
    unsigned short* ko_bf  = p; p += actN;
    unsigned short* vt_bf  = p; p += actN;               // [B,H,32,S]
    unsigned short* hid_bf = p; p += actN;               // attention output

    float* out  = (float*)d_out;                         // [B,L,V]
    float* attn = out + (size_t)Bc * Lc * Vc;            // [B,L,S]

    // ---- zero head-sum accumulator (harness poisons d_out)
    {
        const int n = Bc * Lc * Sc;
        zero_f32<<<dim3((n + 255) / 256), dim3(256), 0, stream>>>(attn, n);
    }

    // ---- one-pass bf16 conversions (amortized: GEMMs never convert)
    {
        const int a4 = (int)(actN / 4), w4 = (int)(wN / 4);
        cvt_bf16<<<dim3((a4 + 255) / 256), 256, 0, stream>>>(q,   q_bf,   a4);
        cvt_bf16<<<dim3((a4 + 255) / 256), 256, 0, stream>>>(k,   k_bf,   a4);
        cvt_bf16<<<dim3((a4 + 255) / 256), 256, 0, stream>>>(v,   v_bf,   a4);
        cvt_bf16<<<dim3((w4 + 255) / 256), 256, 0, stream>>>(wqs, wqs_bf, w4);
        cvt_bf16<<<dim3((w4 + 255) / 256), 256, 0, stream>>>(wqo, wqo_bf, w4);
        cvt_bf16<<<dim3((w4 + 255) / 256), 256, 0, stream>>>(wks, wks_bf, w4);
        cvt_bf16<<<dim3((w4 + 255) / 256), 256, 0, stream>>>(wko, wko_bf, w4);
        cvt_bf16<<<dim3((w4 + 255) / 256), 256, 0, stream>>>(wv,  wv_bf,  w4);
        cvt_bf16<<<dim3((w4 + 255) / 256), 256, 0, stream>>>(fcw, fcw_bf, w4);
    }

    // ---- projections: 16x64 strip per wave, 32 WMMA each
    const dim3 ggrid(nTok / 16, Dc / 64);                // 512 x 4
    const dim3 gblk(32);                                 // one wave32
    gemm_bf<0><<<ggrid, gblk, 0, stream>>>(q_bf, wqs_bf, SCALEc, nullptr, qs_bf);
    gemm_bf<0><<<ggrid, gblk, 0, stream>>>(q_bf, wqo_bf, SCALEc, nullptr, qo_bf);
    gemm_bf<0><<<ggrid, gblk, 0, stream>>>(k_bf, wks_bf, 1.0f,  nullptr, ks_bf);
    gemm_bf<0><<<ggrid, gblk, 0, stream>>>(k_bf, wko_bf, 1.0f,  nullptr, ko_bf);
    gemm_bf<1><<<ggrid, gblk, 0, stream>>>(v_bf, wv_bf,  1.0f,  nullptr, vt_bf);

    // ---- fused attention
    const size_t smem = (size_t)16 * ROWP * sizeof(float) +
                        (size_t)16 * ROWPB * sizeof(unsigned short) +
                        16 * sizeof(int);
    attn_kernel<<<dim3(Lc / 16, Hc, Bc), dim3(32), smem, stream>>>(
        qs_bf, qo_bf, ks_bf, ko_bf, vt_bf, qid, kid, mask, hid_bf, attn);

    // ---- final fc (fp32 out + bias)
    gemm_bf<2><<<ggrid, gblk, 0, stream>>>(hid_bf, fcw_bf, 1.0f, fcb, (void*)out);
}
